// BiMambaEncoderLayer_56710748176552
// MI455X (gfx1250) — matrix operand order repaired and checked by reading, not compiled, not run on hardware
//
#include <hip/hip_runtime.h>

// ---------------------------------------------------------------------------
// BiMamba encoder layer for gfx1250 (MI455X).
// f32 residual stream; all GEMM operands kept as f16 in global memory so tile
// staging is a raw byte-move (async global->LDS DMA with fragment swizzle),
// WMMA f16 x f16 -> f32 accumulate. Wave32 everywhere.
// ---------------------------------------------------------------------------

#define B_SZ    2
#define SEQ_L   2048
#define DMODEL  512
#define DINNER  1024
#define DSTATE  16
#define DTRANK  32
#define BLROWS  (B_SZ * SEQ_L)   // 4096 rows through every GEMM

typedef _Float16 v16h __attribute__((ext_vector_type(16)));
typedef _Float16 h2   __attribute__((ext_vector_type(2)));
typedef float    v8f  __attribute__((ext_vector_type(8)));
typedef int      v4i  __attribute__((ext_vector_type(4)));

// Async global->LDS staging if the toolchain exposes the gfx1250 builtins.
#if defined(__has_builtin)
#  if __has_builtin(__builtin_amdgcn_global_load_async_to_lds_b128)
#    define GEMM_ASYNC 2           // 16-byte DMA chunks
#  elif __has_builtin(__builtin_amdgcn_global_load_async_to_lds_b32)
#    define GEMM_ASYNC 1           // 4-byte DMA chunks
#  endif
#endif
#ifndef GEMM_ASYNC
#  define GEMM_ASYNC 0
#endif

#if GEMM_ASYNC
typedef __attribute__((address_space(1))) int gas_int;   // global (prints as __device__)
typedef __attribute__((address_space(3))) int las_int;   // LDS
typedef __attribute__((address_space(1))) v4i gas_v4i;
typedef __attribute__((address_space(3))) v4i las_v4i;
__device__ __forceinline__ void wait_async0() {
#if __has_builtin(__builtin_amdgcn_s_wait_asynccnt)
    __builtin_amdgcn_s_wait_asynccnt(0);
#else
    asm volatile("s_wait_asynccnt 0x0" ::: "memory");
#endif
}
#endif

// ---------------------------------------------------------------------------
// GEMM: C[M,N] = act(A16[M,K(lda)] * W16[N,K]^T + bias), optional f16 copy.
// ACT: 0 = none, 1 = relu, 2 = softplus. Block tile 128x64, K-step 32,
// 8 wave32 in a 4x2 grid; each wave owns 32x32 of C (4 WMMA accumulators).
// LDS tiles are double-buffered, stored in the WMMA per-lane fragment layout
// (one aligned 32B ds_load per fragment), staged by async DMA.
// ---------------------------------------------------------------------------
template<int ACT, bool HAS_BIAS>
__global__ __launch_bounds__(256)
void gemm_wmma_f16(const _Float16* __restrict__ A, int lda,
                   const _Float16* __restrict__ W,
                   const float* __restrict__ bias,
                   float* __restrict__ C,
                   _Float16* __restrict__ C16,
                   int M, int N, int K)
{
    (void)M;
    // fragment-major tiles: [tile][lane][16 halves]
    __shared__ alignas(32) _Float16 As[2][8 * 512];   // 8 M-tiles of 16x32
    __shared__ alignas(32) _Float16 Bs[2][4 * 512];   // 4 N-tiles of 32x16

    const int tid  = threadIdx.x;
    const int gm0  = blockIdx.y * 128;
    const int gn0  = blockIdx.x * 64;
    const int wv   = tid >> 5;
    const int lane = tid & 31;
    const int mw   = wv >> 1;          // 0..3 : rows mw*32 .. +31
    const int nw   = wv & 1;           // 0..1 : cols nw*32 .. +31
    const int half = lane >> 4;
    const int colL = lane & 15;

    v8f acc00 = {}, acc01 = {}, acc10 = {}, acc11 = {};

    // A tile 128x32 -> fragment-major: element (m,k) of 16x32 sub-tile t goes
    // to lane (m&15)+16*((k>>3)&1), slot (k>>4)*8 + (k&7). Per lane this is
    // two contiguous 16B runs in global (row m, k=h*8.. and k=16+h*8..).
    auto stageA = [&](int buf, int k0) {
#if GEMM_ASYNC == 2
#pragma unroll
        for (int it = 0; it < 2; ++it) {
            const int i  = tid + it * 256;      // 0..511 (8 tiles x 32 lanes x 2)
            const int t  = i >> 6;              // tile
            const int ln = (i >> 1) & 31;       // dest lane
            const int q  = i & 1;               // 16B chunk
            const int m  = t * 16 + (ln & 15);
            const int k  = q * 16 + (ln >> 4) * 8;
            const _Float16* src = A + (size_t)(gm0 + m) * lda + k0 + k;
            __builtin_amdgcn_global_load_async_to_lds_b128(
                (gas_v4i*)src, (las_v4i*)&As[buf][t * 512 + ln * 16 + q * 8], 0, 0);
        }
#elif GEMM_ASYNC == 1
#pragma unroll
        for (int it = 0; it < 8; ++it) {
            const int i = tid + it * 256;          // 0..2047
            const int r = i >> 4;                  // 0..127
            const int c = (i & 15) * 2;            // even k within tile
            const _Float16* src = A + (size_t)(gm0 + r) * lda + k0 + c;
            const int t  = r >> 4;
            const int ln = (r & 15) + (((c >> 3) & 1) << 4);
            const int e  = ((c >> 4) << 3) + (c & 7);
            __builtin_amdgcn_global_load_async_to_lds_b32(
                (gas_int*)src, (las_int*)&As[buf][t * 512 + ln * 16 + e], 0, 0);
        }
#else
        h2 va[8];
#pragma unroll
        for (int it = 0; it < 8; ++it) {
            const int i = tid + it * 256;
            const int r = i >> 4;
            const int c = (i & 15) * 2;
            va[it] = *(const h2*)(A + (size_t)(gm0 + r) * lda + k0 + c);
        }
#pragma unroll
        for (int it = 0; it < 8; ++it) {
            const int i = tid + it * 256;
            const int r = i >> 4;
            const int c = (i & 15) * 2;
            const int t  = r >> 4;
            const int ln = (r & 15) + (((c >> 3) & 1) << 4);
            const int e  = ((c >> 4) << 3) + (c & 7);
            *(h2*)&As[buf][t * 512 + ln * 16 + e] = va[it];
        }
#endif
    };
    // B tile 32x64 -> fragment-major: element (k,n) of sub-tile t goes to
    // lane (n&15)+16*((k>>3)&1), slot (k>>4)*8 + (k&7). Per lane: two 16B
    // runs of W row n (k=h*8.. and k=16+h*8..).
    auto stageB = [&](int buf, int k0) {
#if GEMM_ASYNC == 2
        {
            const int i  = tid;                 // 0..255 (4 tiles x 32 lanes x 2)
            const int t  = i >> 6;              // tile
            const int ln = (i >> 1) & 31;       // dest lane
            const int q  = i & 1;               // 16B chunk
            const int n  = t * 16 + (ln & 15);
            const int k  = q * 16 + (ln >> 4) * 8;
            const _Float16* src = W + (size_t)(gn0 + n) * K + k0 + k;
            __builtin_amdgcn_global_load_async_to_lds_b128(
                (gas_v4i*)src, (las_v4i*)&Bs[buf][t * 512 + ln * 16 + q * 8], 0, 0);
        }
#elif GEMM_ASYNC == 1
#pragma unroll
        for (int it = 0; it < 4; ++it) {
            const int i = tid + it * 256;          // 0..1023
            const int n = i >> 4;                  // 0..63
            const int k = (i & 15) * 2;            // even k within tile
            const _Float16* src = W + (size_t)(gn0 + n) * K + k0 + k;
            const int t  = n >> 4;
            const int ln = (n & 15) + (((k >> 3) & 1) << 4);
            const int e  = ((k >> 4) << 3) + (k & 7);
            __builtin_amdgcn_global_load_async_to_lds_b32(
                (gas_int*)src, (las_int*)&Bs[buf][t * 512 + ln * 16 + e], 0, 0);
        }
#else
        h2 vb[4];
#pragma unroll
        for (int it = 0; it < 4; ++it) {
            const int i = tid + it * 256;
            const int n = i >> 4;
            const int k = (i & 15) * 2;
            vb[it] = *(const h2*)(W + (size_t)(gn0 + n) * K + k0 + k);
        }
#pragma unroll
        for (int it = 0; it < 4; ++it) {
            const int i = tid + it * 256;
            const int n = i >> 4;
            const int k = (i & 15) * 2;
            const int t  = n >> 4;
            const int ln = (n & 15) + (((k >> 3) & 1) << 4);
            const int e  = ((k >> 4) << 3) + (k & 7);
            *(h2*)&Bs[buf][t * 512 + ln * 16 + e] = vb[it];
        }
#endif
    };

    const int nk = K >> 5;
    stageA(0, 0);
    stageB(0, 0);
#if GEMM_ASYNC
    wait_async0();
#endif
    __syncthreads();

    for (int kt = 0; kt < nk; ++kt) {
        const int cur = kt & 1;
        if (kt + 2 < nk) {   // keep HBM/L2 requests in flight (global_prefetch_b8)
            __builtin_prefetch(A + (size_t)(gm0 + (tid >> 1)) * lda + (kt + 2) * 32, 0, 1);
            __builtin_prefetch(W + (size_t)(gn0 + (tid & 63)) * K + (kt + 2) * 32, 0, 1);
        }
        if (kt + 1 < nk) {
            stageA(cur ^ 1, (kt + 1) * 32);
            stageB(cur ^ 1, (kt + 1) * 32);
        }
        const v16h a0 = *(const v16h*)&As[cur][(mw * 2 + 0) * 512 + lane * 16];
        const v16h a1 = *(const v16h*)&As[cur][(mw * 2 + 1) * 512 + lane * 16];
        const v16h b0 = *(const v16h*)&Bs[cur][(nw * 2 + 0) * 512 + lane * 16];
        const v16h b1 = *(const v16h*)&Bs[cur][(nw * 2 + 1) * 512 + lane * 16];
        acc00 = __builtin_amdgcn_wmma_f32_16x16x32_f16(false, a0, false, b0,
                                                       (short)0, acc00, false, false);
        acc01 = __builtin_amdgcn_wmma_f32_16x16x32_f16(false, a0, false, b1,
                                                       (short)0, acc01, false, false);
        acc10 = __builtin_amdgcn_wmma_f32_16x16x32_f16(false, a1, false, b0,
                                                       (short)0, acc10, false, false);
        acc11 = __builtin_amdgcn_wmma_f32_16x16x32_f16(false, a1, false, b1,
                                                       (short)0, acc11, false, false);
#if GEMM_ASYNC
        wait_async0();   // this wave's DMA into buf^1 landed
#endif
        __syncthreads();
    }

    // Epilogue. C layout: VGPR r -> row r + 8*half, col = lane&15.
    const v8f accs[2][2] = { { acc00, acc01 }, { acc10, acc11 } };
#pragma unroll
    for (int i = 0; i < 2; ++i) {
#pragma unroll
        for (int j = 0; j < 2; ++j) {
#pragma unroll
            for (int r = 0; r < 8; ++r) {
                const int m = gm0 + (mw * 2 + i) * 16 + r + half * 8;
                const int n = gn0 + (nw * 2 + j) * 16 + colL;
                float v = accs[i][j][r];
                if (HAS_BIAS) v += bias[n];
                if (ACT == 1)      v = fmaxf(v, 0.0f);
                else if (ACT == 2) v = (v > 20.0f) ? v : __logf(1.0f + __expf(v));
                C[(size_t)m * N + n] = v;
                if (C16) C16[(size_t)m * N + n] = (_Float16)v;
            }
        }
    }
}

// ---------------------------------------------------------------------------
// f32 -> f16 conversion (weights, input activations)
// ---------------------------------------------------------------------------
__global__ __launch_bounds__(256)
void cvt_f16_kernel(const float* __restrict__ in, _Float16* __restrict__ out,
                    int total)
{
    const int i = blockIdx.x * 256 + threadIdx.x;
    if (i >= total) return;
    out[i] = (_Float16)in[i];
}

// ---------------------------------------------------------------------------
// Causal depthwise conv1d (k=4, left pad 3) + SiLU; emits f32 + f16 copies.
// ---------------------------------------------------------------------------
__global__ __launch_bounds__(256)
void conv_silu_kernel(const float* __restrict__ xz,
                      const float* __restrict__ cw,
                      const float* __restrict__ cb,
                      float* __restrict__ out,
                      _Float16* __restrict__ out16, int total)
{
    const int i = blockIdx.x * 256 + threadIdx.x;
    if (i >= total) return;
    const int c  = i & (DINNER - 1);
    const int bl = i >> 10;                 // b*L + l
    const int l  = bl & (SEQ_L - 1);
    const int b  = bl >> 11;
    float acc = cb[c];
#pragma unroll
    for (int j = 0; j < 4; ++j) {
        const int ll = l + j - 3;
        if (ll >= 0)
            acc += cw[c * 4 + j] *
                   xz[((size_t)(b * SEQ_L + ll)) * (2 * DINNER) + c];
    }
    const float s = acc / (1.0f + __expf(-acc));   // SiLU
    out[i]   = s;
    out16[i] = (_Float16)s;
}

// ---------------------------------------------------------------------------
// Selective-scan. One lane per (b, channel, state): 32768 lanes = 1024 waves.
// 16-state reduction via shfl_xor within 16-lane groups (DS permute HW).
// ---------------------------------------------------------------------------
__global__ __launch_bounds__(256)
void scan_kernel(const float* __restrict__ delta,   // [B,L,DINNER] softplus'ed
                 const float* __restrict__ xi,      // [B,L,DINNER] conv+silu
                 const float* __restrict__ dbc,     // [B,L,64]; B@32, C@48
                 const float* __restrict__ xz,      // [B,L,2*DINNER]; z hi half
                 const float* __restrict__ A_log,   // [DINNER,16]
                 const float* __restrict__ Dp,      // [DINNER]
                 float* __restrict__ out,           // [B,L,DINNER]
                 _Float16* __restrict__ out16)
{
    const int gid  = blockIdx.x * 256 + threadIdx.x;  // 0..32767
    const int n    = gid & 15;
    const int slot = gid >> 4;                        // 0..2047
    const int b    = slot >> 10;
    const int c    = slot & (DINNER - 1);

    const float An = -__expf(A_log[c * DSTATE + n]);
    const float Dc = Dp[c];
    float h = 0.0f;
    const size_t base = (size_t)b * SEQ_L;

    for (int l = 0; l < SEQ_L; ++l) {
        const size_t row = base + l;
        const float dlt = delta[row * DINNER + c];
        const float xv  = xi[row * DINNER + c];
        const float Bv  = dbc[row * 64 + DTRANK + n];
        const float Cv  = dbc[row * 64 + DTRANK + DSTATE + n];
        h = __expf(dlt * An) * h + dlt * Bv * xv;
        float y = h * Cv;
        y += __shfl_xor(y, 1, 16);
        y += __shfl_xor(y, 2, 16);
        y += __shfl_xor(y, 4, 16);
        y += __shfl_xor(y, 8, 16);
        if (n == 0) {
            const float zv = xz[row * (2 * DINNER) + DINNER + c];
            const float g  = zv / (1.0f + __expf(-zv));
            const float o  = (y + Dc * xv) * g;
            out[row * DINNER + c]   = o;
            out16[row * DINNER + c] = (_Float16)o;
        }
    }
}

// ---------------------------------------------------------------------------
// out = LayerNorm(a + res) * w + bias, one block per row, D = 512.
// ---------------------------------------------------------------------------
__global__ __launch_bounds__(256)
void ln_add_kernel(const float* __restrict__ a, const float* __restrict__ res,
                   const float* __restrict__ w, const float* __restrict__ bias,
                   float* __restrict__ out, _Float16* __restrict__ out16, int D)
{
    __shared__ float red[256];
    const int row = blockIdx.x;
    const int tid = threadIdx.x;
    const float* pa = a   + (size_t)row * D;
    const float* pr = res + (size_t)row * D;
    const float invD = 1.0f / (float)D;

    float s = 0.0f;
    for (int i = tid; i < D; i += 256) s += pa[i] + pr[i];
    red[tid] = s; __syncthreads();
    for (int off = 128; off > 0; off >>= 1) {
        if (tid < off) red[tid] += red[tid + off];
        __syncthreads();
    }
    const float mean = red[0] * invD;
    __syncthreads();

    float s2 = 0.0f;
    for (int i = tid; i < D; i += 256) {
        const float d = pa[i] + pr[i] - mean;
        s2 += d * d;
    }
    red[tid] = s2; __syncthreads();
    for (int off = 128; off > 0; off >>= 1) {
        if (tid < off) red[tid] += red[tid + off];
        __syncthreads();
    }
    const float rstd = rsqrtf(red[0] * invD + 1e-5f);

    for (int i = tid; i < D; i += 256) {
        const float v = (pa[i] + pr[i] - mean) * rstd * w[i] + bias[i];
        out[(size_t)row * D + i] = v;
        if (out16) out16[(size_t)row * D + i] = (_Float16)v;
    }
}

// ---------------------------------------------------------------------------
// Flip along L within each batch: out[b, L-1-l, :] = in[b, l, :]
// ---------------------------------------------------------------------------
__global__ __launch_bounds__(256)
void flip_kernel(const float* __restrict__ in, float* __restrict__ out,
                 _Float16* __restrict__ out16, int L, int D, int total)
{
    const int i = blockIdx.x * 256 + threadIdx.x;
    if (i >= total) return;
    const int d   = i % D;
    const int bl  = i / D;
    const int l   = bl % L;
    const int b   = bl / L;
    const size_t j = ((size_t)b * L + (L - 1 - l)) * D + d;
    out[j] = in[i];
    if (out16) out16[j] = (_Float16)in[i];
}

__global__ __launch_bounds__(256)
void add_kernel(const float* __restrict__ a, const float* __restrict__ b,
                float* __restrict__ out, int total)
{
    const int i = blockIdx.x * 256 + threadIdx.x;
    if (i >= total) return;
    out[i] = a[i] + b[i];
}

// ---------------------------------------------------------------------------
// Host orchestration
// ---------------------------------------------------------------------------
extern "C" void kernel_launch(void* const* d_in, const int* in_sizes, int n_in,
                              void* d_out, int out_size, void* d_ws, size_t ws_size,
                              hipStream_t stream)
{
    (void)in_sizes; (void)n_in; (void)out_size; (void)ws_size;

    const float* x = (const float*)d_in[0];
    auto P = [&](int base, int off) { return (const float*)d_in[base + off]; };
    const float* ln_w[4] = { (const float*)d_in[19], (const float*)d_in[21],
                             (const float*)d_in[23], (const float*)d_in[25] };
    const float* ln_b[4] = { (const float*)d_in[20], (const float*)d_in[22],
                             (const float*)d_in[24], (const float*)d_in[26] };
    const float* ffn_w1[2] = { (const float*)d_in[27], (const float*)d_in[31] };
    const float* ffn_b1[2] = { (const float*)d_in[28], (const float*)d_in[32] };
    const float* ffn_w2[2] = { (const float*)d_in[29], (const float*)d_in[33] };
    const float* ffn_b2[2] = { (const float*)d_in[30], (const float*)d_in[34] };

    // ---- workspace layout: f32 region then f16 region ----
    float* ws = (float*)d_ws;
    size_t o = 0;
    float* XZ    = ws + o; o += (size_t)BLROWS * 2 * DINNER;
    float* XI    = ws + o; o += (size_t)BLROWS * DINNER;
    float* DBC   = ws + o; o += (size_t)BLROWS * 64;
    float* DELTA = ws + o; o += (size_t)BLROWS * DINNER;
    float* Y     = ws + o; o += (size_t)BLROWS * DINNER;
    float* M1    = ws + o; o += (size_t)BLROWS * DMODEL;
    float* XF    = ws + o; o += (size_t)BLROWS * DMODEL;
    float* H1    = ws + o; o += (size_t)BLROWS * DINNER;
    float* XF2   = ws + o; o += (size_t)BLROWS * DMODEL;
    float* XFLIP = ws + o; o += (size_t)BLROWS * DMODEL;
    float* MFLIP = ws + o; o += (size_t)BLROWS * DMODEL;
    float* XB    = ws + o; o += (size_t)BLROWS * DMODEL;
    float* XB2   = ws + o; o += (size_t)BLROWS * DMODEL;

    _Float16* hb = (_Float16*)(ws + o);
    size_t oh = 0;
    _Float16* X16     = hb + oh; oh += (size_t)BLROWS * DMODEL;
    _Float16* XFLIP16 = hb + oh; oh += (size_t)BLROWS * DMODEL;
    _Float16* XI16    = hb + oh; oh += (size_t)BLROWS * DINNER;
    _Float16* DBC16   = hb + oh; oh += (size_t)BLROWS * 64;
    _Float16* Y16     = hb + oh; oh += (size_t)BLROWS * DINNER;
    _Float16* XF16    = hb + oh; oh += (size_t)BLROWS * DMODEL;
    _Float16* H16     = hb + oh; oh += (size_t)BLROWS * DINNER;
    _Float16* XF216   = hb + oh; oh += (size_t)BLROWS * DMODEL;
    // weight f16 copies: per dir {in_w, xproj_w, dt_w, out_w}, then ffn w1/w2
    _Float16* W16[12];
    const size_t wsz[4] = { (size_t)2 * DINNER * DMODEL, (size_t)64 * DINNER,
                            (size_t)DINNER * DTRANK, (size_t)DMODEL * DINNER };
    for (int d = 0; d < 2; ++d)
        for (int k = 0; k < 4; ++k) { W16[d * 4 + k] = hb + oh; oh += wsz[k]; }
    for (int k = 0; k < 4; ++k)      { W16[8 + k]   = hb + oh; oh += (size_t)DINNER * DMODEL; }

    const int BL = BLROWS;
    auto cvt = [&](const float* in, _Float16* out, size_t tot) {
        cvt_f16_kernel<<<(int)((tot + 255) / 256), 256, 0, stream>>>(in, out, (int)tot);
    };
    auto gemm = [&](const _Float16* A, int lda, const _Float16* W,
                    const float* bias, float* C, _Float16* C16,
                    int M, int N, int K, int act) {
        dim3 grid(N / 64, M / 128);
        if (!bias)
            gemm_wmma_f16<0, false><<<grid, 256, 0, stream>>>(A, lda, W, bias, C, C16, M, N, K);
        else if (act == 0)
            gemm_wmma_f16<0, true><<<grid, 256, 0, stream>>>(A, lda, W, bias, C, C16, M, N, K);
        else if (act == 1)
            gemm_wmma_f16<1, true><<<grid, 256, 0, stream>>>(A, lda, W, bias, C, C16, M, N, K);
        else
            gemm_wmma_f16<2, true><<<grid, 256, 0, stream>>>(A, lda, W, bias, C, C16, M, N, K);
    };

    // ---- one-time (per call) f16 conversions ----
    cvt(x, X16, (size_t)BL * DMODEL);
    for (int d = 0; d < 2; ++d) {
        const int base = 1 + 9 * d;
        cvt(P(base, 0), W16[d * 4 + 0], wsz[0]);   // in_w
        cvt(P(base, 3), W16[d * 4 + 1], wsz[1]);   // xproj_w
        cvt(P(base, 4), W16[d * 4 + 2], wsz[2]);   // dt_w
        cvt(P(base, 8), W16[d * 4 + 3], wsz[3]);   // out_w
    }
    cvt(ffn_w1[0], W16[8],  (size_t)DINNER * DMODEL);
    cvt(ffn_w2[0], W16[9],  (size_t)DINNER * DMODEL);
    cvt(ffn_w1[1], W16[10], (size_t)DINNER * DMODEL);
    cvt(ffn_w2[1], W16[11], (size_t)DINNER * DMODEL);

    auto mamba = [&](const _Float16* xin16, int pbase, const _Float16* const* w16,
                     float* outbuf) {
        gemm(xin16, DMODEL, w16[0], nullptr, XZ, nullptr, BL, 2 * DINNER, DMODEL, 0);
        conv_silu_kernel<<<(BL * DINNER + 255) / 256, 256, 0, stream>>>(
            XZ, P(pbase, 1), P(pbase, 2), XI, XI16, BL * DINNER);
        gemm(XI16, DINNER, w16[1], nullptr, DBC, DBC16, BL, 64, DINNER, 0);
        gemm(DBC16, 64, w16[2], P(pbase, 5), DELTA, nullptr, BL, DINNER, DTRANK, 2);
        scan_kernel<<<(B_SZ * DINNER * DSTATE) / 256, 256, 0, stream>>>(
            DELTA, XI, DBC, XZ, P(pbase, 6), P(pbase, 7), Y, Y16);
        gemm(Y16, DINNER, w16[3], nullptr, outbuf, nullptr, BL, DMODEL, DINNER, 0);
    };

    // ---- forward branch ----
    mamba(X16, 1, &W16[0], M1);
    ln_add_kernel<<<BL, 256, 0, stream>>>(M1, x, ln_w[0], ln_b[0], XF, XF16, DMODEL);
    gemm(XF16, DMODEL, W16[8], ffn_b1[0], H1, H16, BL, DINNER, DMODEL, 1);  // relu
    gemm(H16, DINNER, W16[9], ffn_b2[0], M1, nullptr, BL, DMODEL, DINNER, 0);
    ln_add_kernel<<<BL, 256, 0, stream>>>(M1, XF, ln_w[1], ln_b[1], XF2, XF216, DMODEL);

    // ---- backward branch ----
    flip_kernel<<<(BL * DMODEL + 255) / 256, 256, 0, stream>>>(
        x, XFLIP, XFLIP16, SEQ_L, DMODEL, BL * DMODEL);
    mamba(XFLIP16, 10, &W16[4], M1);
    flip_kernel<<<(BL * DMODEL + 255) / 256, 256, 0, stream>>>(
        M1, MFLIP, nullptr, SEQ_L, DMODEL, BL * DMODEL);
    ln_add_kernel<<<BL, 256, 0, stream>>>(MFLIP, x, ln_w[2], ln_b[2], XB, nullptr, DMODEL);
    // faithful to the reference bug: ffn2 consumes x_f (XF2), not x_b
    gemm(XF216, DMODEL, W16[10], ffn_b1[1], H1, H16, BL, DINNER, DMODEL, 1);
    gemm(H16, DINNER, W16[11], ffn_b2[1], M1, nullptr, BL, DMODEL, DINNER, 0);
    ln_add_kernel<<<BL, 256, 0, stream>>>(M1, XB, ln_w[3], ln_b[3], XB2, nullptr, DMODEL);

    // ---- combine ----
    add_kernel<<<(BL * DMODEL + 255) / 256, 256, 0, stream>>>(
        XF2, XB2, (float*)d_out, BL * DMODEL);
}
